// GATLayer_38826504356577
// MI455X (gfx1250) — matrix operand order repaired and checked by reading, not compiled, not run on hardware
//
#include <hip/hip_runtime.h>
#include <hip/hip_bf16.h>
#include <stdint.h>

typedef float    v2f  __attribute__((ext_vector_type(2)));
typedef float    v8f  __attribute__((ext_vector_type(8)));
typedef _Float16 v8h  __attribute__((ext_vector_type(8)));
typedef _Float16 v16h __attribute__((ext_vector_type(16)));
typedef int      v4i  __attribute__((ext_vector_type(4)));

#define GAT_ALPHA 0.2f
#define GAT_NEGINF -1.0e30f
#define NB 8
#define NN 2048
#define DD 64
#define BSTR 80          // padded LDS row stride (bytes) for a 64B hT tile row -> conflict-free ds reads

#ifndef __has_builtin
#define __has_builtin(x) 0
#endif

#if __has_builtin(__builtin_amdgcn_global_load_async_to_lds_b128)
#define ASYNC_VIA_BUILTIN 1
#else
#define ASYNC_VIA_BUILTIN 0
#endif

#if __has_builtin(__builtin_amdgcn_s_wait_asynccnt)
#define WAIT_ASYNC(n) __builtin_amdgcn_s_wait_asynccnt(n)
#else
#define WAIT_ASYNC(n) asm volatile("s_wait_asynccnt " #n ::: "memory")
#endif

// 16B global -> LDS async copy (ASYNCcnt-tracked, bypasses VGPRs)
__device__ __forceinline__ void cp_async16(const void* g, void* l) {
#if ASYNC_VIA_BUILTIN
    __builtin_amdgcn_global_load_async_to_lds_b128(
        (__attribute__((address_space(1))) v4i*)(void*)g,
        (__attribute__((address_space(3))) v4i*)(void*)l, 0, 0);
#else
    const unsigned loff = (unsigned)(uintptr_t)l;   // low 32 bits of generic ptr = LDS byte offset
    asm volatile("global_load_async_to_lds_b128 %0, %1, off"
                 :: "v"(loff), "v"(g) : "memory");
#endif
}

// -------- Phase 0: pack adj (int32 0/1) into a bitmask via wave32 ballot --------
__global__ __launch_bounds__(256) void gat_pack(
    const int* __restrict__ adj, unsigned* __restrict__ abits)
{
    const int lane = threadIdx.x & 31;
    const int row  = blockIdx.x * 8 + (threadIdx.x >> 5);   // 0..2047
    const int* ar  = adj + (size_t)row * NN;
    for (int it = 0; it < NN / 32; ++it) {
        const unsigned long long bm = __ballot(ar[it * 32 + lane] > 0);
        if (lane == 0) abits[row * (NN / 32) + it] = (unsigned)bm;
    }
}

// -------- Phase 1: h = x@W via V_WMMA_F32_16X16X4_F32; s=h.a1, t=h.a2; hT f16 --------
__global__ __launch_bounds__(128) void gat_phase1(
    const float* __restrict__ x, const float* __restrict__ W,
    const float* __restrict__ a, _Float16* __restrict__ hT,
    float* __restrict__ s, float* __restrict__ t)
{
    const int lane = threadIdx.x & 31;
    const int wid  = threadIdx.x >> 5;
    const int wt   = blockIdx.x * 4 + wid;        // 0..1023 = 8 batches * 128 tiles
    const int b    = wt >> 7;
    const int i0   = (wt & 127) << 4;
    const int m    = lane & 15;
    const int hh   = lane >> 4;

    const float* xrow = x + (size_t)(b * NN + i0 + m) * DD;

    v8f acc[4] = {};
    #pragma unroll 4
    for (int k0 = 0; k0 < DD; k0 += 4) {
        const int ka = k0 + 2 * hh;
        v2f A;
        A.x = xrow[ka];
        A.y = xrow[ka + 1];
        #pragma unroll
        for (int c = 0; c < 4; ++c) {
            const int n = c * 16 + m;
            v2f Bv;
            Bv.x = W[ka * DD + n];
            Bv.y = W[(ka + 1) * DD + n];
            acc[c] = __builtin_amdgcn_wmma_f32_16x16x4_f32(
                false, A, false, Bv, (short)0, acc[c], false, false);
        }
    }

    const float* a1 = a;
    const float* a2 = a + DD;
    float a1v[4], a2v[4];
    #pragma unroll
    for (int c = 0; c < 4; ++c) { a1v[c] = a1[c * 16 + m]; a2v[c] = a2[c * 16 + m]; }

    #pragma unroll
    for (int g = 0; g < 8; ++g) {
        float sp = 0.0f, tp = 0.0f;
        #pragma unroll
        for (int c = 0; c < 4; ++c) {
            sp += acc[c][g] * a1v[c];
            tp += acc[c][g] * a2v[c];
        }
        #pragma unroll
        for (int mask = 8; mask >= 1; mask >>= 1) {
            sp += __shfl_xor(sp, mask, 32);
            tp += __shfl_xor(tp, mask, 32);
        }
        if (m == 0) {
            const int row = i0 + 8 * hh + g;
            s[b * NN + row] = sp;
            t[b * NN + row] = tp;
        }
    }

    #pragma unroll
    for (int c = 0; c < 4; ++c)
        #pragma unroll
        for (int g = 0; g < 8; ++g)
            hT[(size_t)(b * DD + c * 16 + m) * NN + (i0 + 8 * hh + g)] =
                (_Float16)acc[c][g];
}

// -------- Phase 2: per-row stats (max, 1/sum). Scores kept in regs; 1 exp/element --------
__global__ __launch_bounds__(256) void gat_phase2(
    const float* __restrict__ s, const float* __restrict__ t,
    const unsigned* __restrict__ abits, float* __restrict__ rowmax,
    float* __restrict__ rowinv)
{
    const int lane = threadIdx.x & 31;
    const int wg   = blockIdx.x * 8 + (threadIdx.x >> 5);   // 0..16383 = b*2048 + i
    const int i    = wg & (NN - 1);
    const int b    = wg >> 11;

    const float     si = s[wg];
    const float*    tb = t + b * NN;
    const unsigned* ar = abits + i * (NN / 32);

    float ev[NN / 32];
    float mrun = -3.0e38f;
    #pragma unroll
    for (int it = 0; it < NN / 32; ++it) {
        const unsigned w = ar[it];                 // same word for all 32 lanes (broadcast)
        float e = si + tb[it * 32 + lane];         // coalesced
        e = fmaxf(e, GAT_ALPHA * e);               // LeakyReLU
        ev[it] = ((w >> lane) & 1u) ? e : GAT_NEGINF;
        mrun = fmaxf(mrun, ev[it]);
    }
    #pragma unroll
    for (int mask = 16; mask >= 1; mask >>= 1)
        mrun = fmaxf(mrun, __shfl_xor(mrun, mask, 32));

    float ssum = 0.0f;
    #pragma unroll
    for (int it = 0; it < NN / 32; ++it)
        ssum += __expf(ev[it] - mrun);
    #pragma unroll
    for (int mask = 16; mask >= 1; mask >>= 1)
        ssum += __shfl_xor(ssum, mask, 32);

    if (lane == 0) {
        rowmax[wg] = mrun;
        rowinv[wg] = 1.0f / ssum;     // all-masked row -> sum = N -> uniform, matches ref
    }
}

// -------- Phase 3: out = P @ h via V_WMMA_F32_16X16X32_F16 --------
// hT j-tiles async-staged into LDS (double buffered), shared by the 4 waves of a block.
// A(16x32 f16) lane layout: lane L: row m=L&15; slots 0..7 -> K=8*hh+0..7,
// slots 8..15 -> K=16+8*hh+0..7. B(32x16 f16): lane col n=L&15, slots 0..7 -> K=8*hh+0..7, slots 8..15 -> K=16+8*hh+0..7 (as A).
__global__ __launch_bounds__(128) void gat_phase3(
    const float* __restrict__ s, const float* __restrict__ t,
    const unsigned* __restrict__ abits, const _Float16* __restrict__ hT,
    const float* __restrict__ rowmax, const float* __restrict__ rowinv,
    float* __restrict__ out)
{
    __shared__ alignas(16) float tlds[NN];                 // 8 KB: t[b][:]
    __shared__ uint4 bbuf[2][DD * BSTR / 16];              // 2 x 5 KB padded hT tiles

    const int tid  = threadIdx.x;
    const int lane = tid & 31;
    const int wid  = tid >> 5;
    const int wt   = blockIdx.x * 4 + wid;        // 4 waves of a block: same batch
    const int b    = wt >> 7;
    const int i0   = (wt & 127) << 4;
    const int m    = lane & 15;
    const int hh   = lane >> 4;
    const int i    = i0 + m;

    const float     si  = s[b * NN + i];
    const float     mx  = rowmax[b * NN + i];
    const float     inv = rowinv[b * NN + i];
    const unsigned* ar  = abits + i * (NN / 32);
    const _Float16* hTb = hT + (size_t)b * DD * NN;

    // stage t[b][:] into LDS: 512 16B chunks over 128 threads
    #pragma unroll
    for (int k = 0; k < 4; ++k) {
        const int idx = tid + k * 128;
        cp_async16(t + b * NN + idx * 4, &tlds[idx * 4]);
    }
    // B-tile stage: 256 16B chunks (64 rows x 64B, padded to BSTR) over 128 threads
    auto prefetch_btile = [&](int j0, int buf) {
        #pragma unroll
        for (int k = 0; k < 2; ++k) {
            const int c = tid + k * 128;
            const int d = c >> 2, q = c & 3;
            cp_async16(hTb + (size_t)d * NN + j0 + q * 8,
                       (char*)bbuf[buf] + d * BSTR + q * 16);
        }
    };
    prefetch_btile(0, 0);

    v8f acc[4] = {};

    auto pcalc = [&](float tv, unsigned bit) -> _Float16 {
        float e = si + tv;
        e = fmaxf(e, GAT_ALPHA * e);               // LeakyReLU
        e = bit ? e : GAT_NEGINF;                  // adjacency mask
        return (_Float16)__expf(e - mx);           // unnormalized; 1/sum applied at the end
    };

    for (int kt = 0; kt < NN / 32; ++kt) {
        const int j0 = kt * 32;
        if (kt + 1 < NN / 32) {
            prefetch_btile(j0 + 32, (kt + 1) & 1);
            WAIT_ASYNC(2);                         // current tile (and t) landed in LDS
        } else {
            WAIT_ASYNC(0);
        }
        __syncthreads();                           // all waves' copies visible

        if ((kt & 7) == 0) __builtin_prefetch(ar + kt + 8, 0, 1);
        const unsigned w  = ar[kt];
        const unsigned uA = (w >> (8 * hh)) & 0xffu;
        const unsigned uB = (w >> (8 * hh + 16)) & 0xffu;
        const int jA = j0 + 8 * hh;

        const float4 t0 = *(const float4*)&tlds[jA];
        const float4 t1 = *(const float4*)&tlds[jA + 4];
        const float4 t2 = *(const float4*)&tlds[jA + 16];
        const float4 t3 = *(const float4*)&tlds[jA + 20];

        v16h A;
        A[0]  = pcalc(t0.x, uA & 1u);   A[1]  = pcalc(t0.y, uA & 2u);
        A[2]  = pcalc(t0.z, uA & 4u);   A[3]  = pcalc(t0.w, uA & 8u);
        A[4]  = pcalc(t1.x, uA & 16u);  A[5]  = pcalc(t1.y, uA & 32u);
        A[6]  = pcalc(t1.z, uA & 64u);  A[7]  = pcalc(t1.w, uA & 128u);
        A[8]  = pcalc(t2.x, uB & 1u);   A[9]  = pcalc(t2.y, uB & 2u);
        A[10] = pcalc(t2.z, uB & 4u);   A[11] = pcalc(t2.w, uB & 8u);
        A[12] = pcalc(t3.x, uB & 16u);  A[13] = pcalc(t3.y, uB & 32u);
        A[14] = pcalc(t3.z, uB & 64u);  A[15] = pcalc(t3.w, uB & 128u);

        const char* bb = (const char*)bbuf[kt & 1];
        #pragma unroll
        for (int c = 0; c < 4; ++c) {
            const char* base = bb + (c * 16 + m) * BSTR + hh * 16;
            const v8h lo = *(const v8h*)base;            // conflict-free ds_load_b128
            const v8h hi = *(const v8h*)(base + 32);
            const v16h Bv = __builtin_shufflevector(lo, hi,
                0, 1, 2, 3, 4, 5, 6, 7, 8, 9, 10, 11, 12, 13, 14, 15);
            acc[c] = __builtin_amdgcn_wmma_f32_16x16x32_f16(
                false, A, false, Bv, (short)0, acc[c], false, false);
        }
        __syncthreads();                           // tile consumed; its buffer reusable
    }

    // C/D layout: lane L holds rows 8*hh+g, col c*16+m. Apply 1/sum per row, coalesced stores.
    float* ob = out + (size_t)(b * NN + i0 + 8 * hh) * DD + m;
    #pragma unroll
    for (int g = 0; g < 8; ++g) {
        const float sc = __shfl(inv, 8 * hh + g, 32);    // lane r<16 holds row r's 1/sum
        #pragma unroll
        for (int c = 0; c < 4; ++c)
            ob[(size_t)g * DD + c * 16] = acc[c][g] * sc;
    }
}

extern "C" void kernel_launch(void* const* d_in, const int* in_sizes, int n_in,
                              void* d_out, int out_size, void* d_ws, size_t ws_size,
                              hipStream_t stream)
{
    const float* x   = (const float*)d_in[0];   // [8,2048,64] f32
    const int*   adj = (const int*)d_in[1];     // [2048,2048] i32
    const float* W   = (const float*)d_in[2];   // [64,64] f32
    const float* a   = (const float*)d_in[3];   // [128] f32
    float*       out = (float*)d_out;           // [8,2048,64] f32

    char* ws = (char*)d_ws;
    _Float16* hT    = (_Float16*)ws;                                   // 2 MB
    float*    s     = (float*)(ws + (size_t)NB * DD * NN * sizeof(_Float16));
    float*    t     = s   + NB * NN;
    float*    rmx   = t   + NB * NN;
    float*    rin   = rmx + NB * NN;
    unsigned* abits = (unsigned*)(rin + NB * NN);                      // 512 KB

    gat_pack  <<<NN / 8, 256, 0, stream>>>(adj, abits);
    gat_phase1<<<256, 128, 0, stream>>>(x, W, a, hT, s, t);
    gat_phase2<<<2048, 256, 0, stream>>>(s, t, abits, rmx, rin);
    gat_phase3<<<256, 128, 0, stream>>>(s, t, abits, hT, rmx, rin, out);
}
